// BiLSTM_25907242730057
// MI455X (gfx1250) — matrix operand order repaired and checked by reading, not compiled, not run on hardware
//
#include <hip/hip_runtime.h>
#include <hip/hip_bf16.h>
#include <stdint.h>

#define S_LEN 4096
#define H_DIM 1024
#define I_DIM 1024
#define O_DIM 1024
#define G_DIM 4096   // 4*H

typedef __attribute__((ext_vector_type(16))) __bf16 v16bf;
typedef __attribute__((ext_vector_type(8)))  float  v8f;

union Frag { v16bf v; uint4 q[2]; };

__device__ __forceinline__ float bf_lo(unsigned u){ return __uint_as_float(u << 16); }
__device__ __forceinline__ float bf_hi(unsigned u){ return __uint_as_float(u & 0xffff0000u); }

// --------------------------------------------------------------------------
// gfx1250 async global->LDS DMA (ASYNCcnt-tracked), guarded by __has_builtin.
// Builtin signature (from compiler diagnostic): param0 = v4i* in AS1
// ("__device__"), so we pass typed v4i pointers. AS casts go through
// integers: generic LDS ptr low 32 bits == LDS offset, global VA == AS1.
// --------------------------------------------------------------------------
#if __has_builtin(__builtin_amdgcn_global_load_async_to_lds_b128)
#define ASYNC_LDS 1
typedef int v4i_vs __attribute__((vector_size(16)));
typedef __attribute__((address_space(1))) v4i_vs as1_v4i;
typedef __attribute__((address_space(3))) v4i_vs as3_v4i;
__device__ __forceinline__ as3_v4i* to_lds(void* p) {
    return (as3_v4i*)(unsigned)(uintptr_t)p;
}
__device__ __forceinline__ as1_v4i* to_glb(const void* p) {
    return (as1_v4i*)(uintptr_t)p;
}
__device__ __forceinline__ void wait_async0() {
#if __has_builtin(__builtin_amdgcn_s_wait_asynccnt)
    __builtin_amdgcn_s_wait_asynccnt(0);
#else
    asm volatile("s_wait_asynccnt 0" ::: "memory");
#endif
}
#endif

// ---------------------------------------------------------------------------
// Init: zero h state + barrier counters (ws is poisoned 0xAA by harness).
// ---------------------------------------------------------------------------
__global__ void init_state(float* h_cur, unsigned* ctr) {
    int t = blockIdx.x * blockDim.x + threadIdx.x;
    if (t < 2 * H_DIM) h_cur[t] = 0.0f;
    if (t < 2) ctr[t] = 0u;
}

// ---------------------------------------------------------------------------
// Pack fp32 -> bf16 (row-major, used for X).
// ---------------------------------------------------------------------------
__global__ void pack_x_bf16(const float* __restrict__ X, __bf16* __restrict__ Xb, int n) {
    int i = blockIdx.x * blockDim.x + threadIdx.x;
    if (i < n) Xb[i] = (__bf16)X[i];
}

// ---------------------------------------------------------------------------
// Pack W (N rows x ldw cols, take cols [colbase, colbase+K)) into WMMA
// B-fragment tiles: tile (nt,kt) is 32x16 bf16 = 512 elems contiguous;
// lane l holds 16 bf16 (K = kt*32 + (j<8?0:16) + (l<16?0:8) + (j&7), N = nt*16 + (l&15)).
// ---------------------------------------------------------------------------
__global__ void pack_B_frag(const float* __restrict__ W, __bf16* __restrict__ Bp,
                            int N, int K, int ldw, int colbase) {
    size_t tid = (size_t)blockIdx.x * blockDim.x + threadIdx.x;
    size_t total = (size_t)N * K;
    if (tid >= total) return;
    int j = (int)(tid & 15);
    int l = (int)((tid >> 4) & 31);
    size_t rest = tid >> 9;
    int KT = K >> 5;
    int kt = (int)(rest % KT);
    int nt = (int)(rest / KT);
    int n = nt * 16 + (l & 15);
    int k = kt * 32 + ((j < 8) ? 0 : 16) + ((l < 16) ? 0 : 8) + (j & 7);
    Bp[tid] = (__bf16)W[(size_t)n * ldw + colbase + k];
}

// ---------------------------------------------------------------------------
// Pack the recurrent weights W[:, 0:H] (both directions) to bf16 row-major:
// Wh[d][r][k], r in [0,4096), k in [0,1024).
// ---------------------------------------------------------------------------
__global__ void pack_Wh_bf16(const float* __restrict__ Wl, const float* __restrict__ Wr,
                             __bf16* __restrict__ Wh) {
    size_t tid = (size_t)blockIdx.x * blockDim.x + threadIdx.x;
    size_t total = (size_t)2 * G_DIM * H_DIM;
    if (tid >= total) return;
    int k = (int)(tid & (H_DIM - 1));
    int r = (int)((tid >> 10) & (G_DIM - 1));
    int d = (int)(tid >> 22);
    const float* W = d ? Wr : Wl;
    Wh[tid] = (__bf16)W[(size_t)r * (H_DIM + I_DIM) + k];
}

// ---------------------------------------------------------------------------
// WMMA bf16 GEMM with bias: C(MxN,f32) = A(MxK,bf16 row-major) * B + bias[n].
// B pre-packed by pack_B_frag. One wave computes a 16x64 output strip.
// Two-deep register ping-pong: fragments for kt+1 are loaded while WMMAs for
// kt issue, so s_wait_loadcnt covers loads that are a full K-step old.
// K multiple of 64 (KT even) in all uses here.
// ---------------------------------------------------------------------------
__global__ void __launch_bounds__(256) gemm_bf16_bias(
    const __bf16* __restrict__ A, const __bf16* __restrict__ Bp,
    const float* __restrict__ bias, float* __restrict__ C,
    int M, int N, int K)
{
    int KT = K >> 5;
    int lane = threadIdx.x & 31;
    int wid  = (blockIdx.x << 3) + (threadIdx.x >> 5);
    int nSuper = N >> 6;
    int total = (M >> 4) * nSuper;
    if (wid >= total) return;
    int mt  = wid / nSuper;
    int ntb = wid % nSuper;

    int m    = lane & 15;
    int koff = (lane < 16) ? 0 : 8;          // A fragment K interleave (16-bit A layout)
    const __bf16* arow  = A + (size_t)(mt * 16 + m) * K + koff;
    const __bf16* Bbase = Bp + ((size_t)(ntb * 4) * KT) * 512 + lane * 16;

    v8f acc[4];
#pragma unroll
    for (int b = 0; b < 4; ++b) acc[b] = (v8f){0,0,0,0,0,0,0,0};

    auto loadA = [&](Frag& a, int kt) {
        const uint4* ap = (const uint4*)(arow + kt * 32);
        a.q[0] = ap[0];          // K = koff .. koff+7
        a.q[1] = ap[2];          // K = koff+16 .. koff+23
    };
    auto loadB = [&](Frag* bb, int kt) {
#pragma unroll
        for (int b = 0; b < 4; ++b) {
            const uint4* bp = (const uint4*)(Bbase + ((size_t)b * KT + kt) * 512);
            bb[b].q[0] = bp[0];
            bb[b].q[1] = bp[1];
        }
    };
    auto mma4 = [&](Frag& a, Frag* bb) {
#pragma unroll
        for (int b = 0; b < 4; ++b)
            acc[b] = __builtin_amdgcn_wmma_f32_16x16x32_bf16(
                false, a.v, false, bb[b].v, (short)0, acc[b], false, false);
    };

    Frag a0, a1, b0[4], b1[4];
    loadA(a0, 0);
    loadB(b0, 0);
    for (int kt = 0; kt < KT; kt += 2) {
        loadA(a1, kt + 1);
        loadB(b1, kt + 1);
        __builtin_prefetch((const void*)(arow + (kt + 2) * 32), 0, 1);  // speculative
        mma4(a0, b0);
        if (kt + 2 < KT) {
            loadA(a0, kt + 2);
            loadB(b0, kt + 2);
        }
        mma4(a1, b1);
    }

    int rbase = mt * 16 + ((lane < 16) ? 0 : 8);
#pragma unroll
    for (int b = 0; b < 4; ++b) {
        int n = ntb * 64 + b * 16 + (lane & 15);
        float bv = bias[n];
#pragma unroll
        for (int i = 0; i < 8; ++i)
            C[(size_t)(rbase + i) * N + n] = acc[b][i] + bv;
    }
}

// ---------------------------------------------------------------------------
// Recurrence: persistent grid, 64 WGs per direction (grid = 128 blocks of 256).
// Each WG owns 16 hidden units -> 64 rows of W_h (4 gates x 16), kept bf16 in
// LDS (padded rows, ~129 KB) for the whole sequence. Per timestep:
//   async-DMA h (4 KB) to LDS -> 64x1024 matvec (4 partials/row) -> gates ->
//   update c,h -> device-scope store of h + bf16 store into hs -> atomic
//   arrive; monotonic counter barrier synchronizes the direction's 64 WGs.
// ---------------------------------------------------------------------------
__global__ void __launch_bounds__(256) bilstm_rec(
    const __bf16* __restrict__ Wh,    // [2][4096][1024] bf16
    const float*  __restrict__ Gx,    // [2][S][4096] f32 (bias included)
    float*        __restrict__ h_cur, // [2][1024] f32
    __bf16*       __restrict__ hs,    // [S][2048] bf16
    unsigned*     __restrict__ ctr)   // [2]
{
    extern __shared__ char smem[];
    const int ROWB = 2064;                        // 1024 bf16 + 8 pad (bank skew)
    float* h_lds = (float*)(smem + 64 * ROWB);    // 1024 f32
    float* pr    = h_lds + H_DIM;                 // 256 partials
    float* g_lds = pr + 256;                      // 64 gate pre-activations

    int tid = threadIdx.x;
    int dir = blockIdx.x >> 6;
    int wg  = blockIdx.x & 63;
    int j0  = wg * 16;

    // Load this WG's 64 rows of W_h into LDS (async DMA if available).
#ifdef ASYNC_LDS
    for (int idx = tid; idx < 64 * 128; idx += 256) {
        int r = idx >> 7, c16 = idx & 127;
        int g8 = r >> 4, jr = r & 15;
        const __bf16* gsrc = Wh + ((size_t)dir * G_DIM + g8 * H_DIM + j0 + jr) * H_DIM + c16 * 8;
        __builtin_amdgcn_global_load_async_to_lds_b128(
            to_glb((const void*)gsrc), to_lds((void*)(smem + r * ROWB + c16 * 16)), 0, 0);
    }
    wait_async0();
#else
    for (int idx = tid; idx < 64 * 128; idx += 256) {
        int r = idx >> 7, c16 = idx & 127;
        int g8 = r >> 4, jr = r & 15;
        const uint4* src = (const uint4*)(Wh + ((size_t)dir * G_DIM + g8 * H_DIM + j0 + jr) * H_DIM) + c16;
        *((uint4*)(smem + r * ROWB) + c16) = *src;
    }
#endif
    __syncthreads();

    int r = tid >> 2, part = tid & 3;
    const uint4* wrow = (const uint4*)(smem + r * ROWB + part * 512);
    const float* hh   = h_lds + part * 256;
    int gate = tid >> 4;                          // valid when tid < 64
    float c_reg = 0.0f;

    for (int t = 0; t < S_LEN; ++t) {
        // Wait until all 64 WGs of this direction published h for step t.
        if (tid == 0 && t > 0) {
            unsigned target = 64u * (unsigned)t;
            while (__hip_atomic_load(ctr + dir, __ATOMIC_ACQUIRE, __HIP_MEMORY_SCOPE_AGENT) < target)
                __builtin_amdgcn_s_sleep(2);
        }
        __syncthreads();

        // Stage h_{t-1} into LDS.
#ifdef ASYNC_LDS
        __builtin_amdgcn_global_load_async_to_lds_b128(
            to_glb((const void*)(h_cur + dir * H_DIM + tid * 4)),
            to_lds((void*)(h_lds + tid * 4)), 0, 0);
        wait_async0();
#else
        {
            int base = tid * 4;
            h_lds[base + 0] = __hip_atomic_load(h_cur + dir * H_DIM + base + 0, __ATOMIC_RELAXED, __HIP_MEMORY_SCOPE_AGENT);
            h_lds[base + 1] = __hip_atomic_load(h_cur + dir * H_DIM + base + 1, __ATOMIC_RELAXED, __HIP_MEMORY_SCOPE_AGENT);
            h_lds[base + 2] = __hip_atomic_load(h_cur + dir * H_DIM + base + 2, __ATOMIC_RELAXED, __HIP_MEMORY_SCOPE_AGENT);
            h_lds[base + 3] = __hip_atomic_load(h_cur + dir * H_DIM + base + 3, __ATOMIC_RELAXED, __HIP_MEMORY_SCOPE_AGENT);
        }
#endif
        __syncthreads();

        // 64 rows x 1024 matvec: 4 threads per row, 256 MACs each.
        float sum = 0.0f;
#pragma unroll 4
        for (int it = 0; it < 32; ++it) {
            uint4 w = wrow[it];
            const float* hp = hh + it * 8;
            sum += bf_lo(w.x) * hp[0] + bf_hi(w.x) * hp[1];
            sum += bf_lo(w.y) * hp[2] + bf_hi(w.y) * hp[3];
            sum += bf_lo(w.z) * hp[4] + bf_hi(w.z) * hp[5];
            sum += bf_lo(w.w) * hp[6] + bf_hi(w.w) * hp[7];
        }
        pr[tid] = sum;     // tid == r*4 + part
        __syncthreads();

        int row = (dir == 0) ? t : (S_LEN - 1 - t);
        if (tid < 64) {
            float g = pr[tid * 4] + pr[tid * 4 + 1] + pr[tid * 4 + 2] + pr[tid * 4 + 3]
                    + Gx[((size_t)dir * S_LEN + row) * G_DIM + gate * H_DIM + j0 + (tid & 15)];
            g_lds[tid] = g;
        }
        __syncthreads();

        if (tid < 16) {
            float gf = g_lds[tid];
            float gi = g_lds[16 + tid];
            float gc = g_lds[32 + tid];
            float go = g_lds[48 + tid];
            float f  = 1.0f / (1.0f + __expf(-gf));
            float i  = 1.0f / (1.0f + __expf(-gi));
            float ct = tanhf(gc);
            float o  = 1.0f / (1.0f + __expf(-go));
            c_reg = f * c_reg + i * ct;
            float hn = o * tanhf(c_reg);
            __hip_atomic_store(h_cur + dir * H_DIM + j0 + tid, hn, __ATOMIC_RELAXED, __HIP_MEMORY_SCOPE_AGENT);
            hs[(size_t)row * (2 * H_DIM) + dir * H_DIM + j0 + tid] = (__bf16)hn;
        }
        __syncthreads();

        if (tid == 0) {
            __threadfence();   // release h stores to device scope
            __hip_atomic_fetch_add(ctr + dir, 1u, __ATOMIC_RELEASE, __HIP_MEMORY_SCOPE_AGENT);
        }
    }
}

// ---------------------------------------------------------------------------
// Host launcher
// ---------------------------------------------------------------------------
extern "C" void kernel_launch(void* const* d_in, const int* in_sizes, int n_in,
                              void* d_out, int out_size, void* d_ws, size_t ws_size,
                              hipStream_t stream) {
    (void)in_sizes; (void)n_in; (void)out_size; (void)ws_size;

    const float* X   = (const float*)d_in[0];
    const float* W_l = (const float*)d_in[1];
    const float* b_l = (const float*)d_in[2];
    const float* W_r = (const float*)d_in[3];
    const float* b_r = (const float*)d_in[4];
    const float* W_y = (const float*)d_in[5];
    const float* b_y = (const float*)d_in[6];
    float* out = (float*)d_out;

    char* ws = (char*)d_ws;
    const size_t MB = 1024ull * 1024ull;
    __bf16*   Xbf = (__bf16*)(ws);                 //   8 MB: X bf16
    __bf16*   BxL = (__bf16*)(ws + 8 * MB);        //   8 MB: packed W_x left
    __bf16*   BxR = (__bf16*)(ws + 16 * MB);       //   8 MB: packed W_x right
    __bf16*   Whb = (__bf16*)(ws + 24 * MB);       //  16 MB: W_h both dirs bf16
    __bf16*   By  = (__bf16*)(ws + 40 * MB);       //   4 MB: packed W_y
    float*    GxL = (float*)(ws + 44 * MB);        // 128 MB: Gx both dirs (contig)
    __bf16*   hsb = (__bf16*)(ws + 172 * MB);      //  16 MB: hs bf16 [S][2H]
    float*    hc  = (float*)(ws + 188 * MB);       //   8 KB: h state [2][H]
    unsigned* ctr = (unsigned*)(ws + 188 * MB + 16384);
    float*    GxR = GxL + (size_t)S_LEN * G_DIM;

    // 1. zero recurrence state + barrier counters
    init_state<<<8, 256, 0, stream>>>(hc, ctr);

    // 2. pack inputs/weights to bf16 / fragment layout
    pack_x_bf16<<<(S_LEN * I_DIM) / 256, 256, 0, stream>>>(X, Xbf, S_LEN * I_DIM);
    pack_B_frag<<<(G_DIM * I_DIM) / 256, 256, 0, stream>>>(W_l, BxL, G_DIM, I_DIM, H_DIM + I_DIM, H_DIM);
    pack_B_frag<<<(G_DIM * I_DIM) / 256, 256, 0, stream>>>(W_r, BxR, G_DIM, I_DIM, H_DIM + I_DIM, H_DIM);
    pack_Wh_bf16<<<(2 * G_DIM * H_DIM) / 256, 256, 0, stream>>>(W_l, W_r, Whb);
    pack_B_frag<<<(O_DIM * 2 * H_DIM) / 256, 256, 0, stream>>>(W_y, By, O_DIM, 2 * H_DIM, 2 * H_DIM, 0);

    // 3. WMMA GEMMs: Gx_d = X @ W_x_d^T + b_d   (M=4096, N=4096, K=1024)
    gemm_bf16_bias<<<2048, 256, 0, stream>>>(Xbf, BxL, b_l, GxL, S_LEN, G_DIM, I_DIM);
    gemm_bf16_bias<<<2048, 256, 0, stream>>>(Xbf, BxR, b_r, GxR, S_LEN, G_DIM, I_DIM);

    // 4. sequential recurrence, both directions in one persistent launch
    bilstm_rec<<<128, 256, 64 * 2064 + 4096 + 1024 + 256, stream>>>(Whb, GxL, hc, hsb, ctr);

    // 5. WMMA output projection: out = hs @ W_y^T + b_y  (M=4096, N=1024, K=2048)
    gemm_bf16_bias<<<512, 256, 0, stream>>>(hsb, By, b_y, out, S_LEN, O_DIM, 2 * H_DIM);
}
